// VGAE_32409823216073
// MI455X (gfx1250) — matrix operand order repaired and checked by reading, not compiled, not run on hardware
//
#include <hip/hip_runtime.h>

typedef __attribute__((ext_vector_type(16))) __bf16 v16bf;
typedef __attribute__((ext_vector_type(8)))  float  v8f;

#define TK  32   // K step per WMMA
#define TKP 40   // padded LDS row stride (shorts): 80B -> bank-conflict-free, 16B aligned
#define TM  64   // workgroup M tile
#define TN  64   // workgroup N tile

// Split fp32 into hi/lo bf16 (x ~= hi + lo, ~16 mantissa bits total).
__device__ __forceinline__ void split2bf16(float x, unsigned short& hi, unsigned short& lo) {
    unsigned int u = __float_as_uint(x);
    hi = (unsigned short)(u >> 16);                               // truncate -> hi
    float r = x - __uint_as_float(u & 0xFFFF0000u);               // exact residual
    lo = (unsigned short)((__float_as_uint(r) + 0x8000u) >> 16);  // RN -> lo
}

struct BF4 { unsigned short h[4], l[4]; };
__device__ __forceinline__ BF4 split4(float4 v) {
    BF4 r;
    split2bf16(v.x, r.h[0], r.l[0]);
    split2bf16(v.y, r.h[1], r.l[1]);
    split2bf16(v.z, r.h[2], r.l[2]);
    split2bf16(v.w, r.h[3], r.l[3]);
    return r;
}
__device__ __forceinline__ void store4(unsigned short* p, const unsigned short s[4]) {
    uint2 u;
    u.x = (unsigned int)s[0] | ((unsigned int)s[1] << 16);
    u.y = (unsigned int)s[2] | ((unsigned int)s[3] << 16);
    *(uint2*)p = u;   // 8B-aligned packed LDS store
}

// C[M,N] (+)= A[M,K] * B[K,N] via bf16x3 emulated fp32 (3x v_wmma_f32_16x16x32_bf16).
// Grid (M/64, N/64, kChunks). kChunks>1 -> each chunk writes partial slab
// P + z*sliceMN (reduced later in fixed order; fully deterministic, no atomics).
// bTrans: B element (k,n) at B[n*ldb + k] (used for mu @ mu^T).
__global__ __launch_bounds__(256) void gemm_bf16x3(
    const float* __restrict__ A, int lda,
    const float* __restrict__ B, int ldb, int bTrans,
    float* __restrict__ C, int ldc,
    float* __restrict__ P, long sliceMN, int kLen)
{
    __shared__ unsigned short AsHi[TM * TKP];
    __shared__ unsigned short AsLo[TM * TKP];
    __shared__ unsigned short BsHi[TN * TKP];   // stored [n][k] -> K-contiguous
    __shared__ unsigned short BsLo[TN * TKP];

    const int t    = threadIdx.x;
    const int lane = t & 31;
    const int wave = t >> 5;
    const int lh   = lane >> 4;          // lane half
    const int lm   = lane & 15;
    const int mt   = wave >> 1;          // m-tile 0..3
    const int nt0  = (wave & 1) * 2;     // n-tiles nt0, nt0+1

    const long m0     = (long)blockIdx.x * TM;
    const long n0     = (long)blockIdx.y * TN;
    const int  kStart = blockIdx.z * kLen;

    float* Ct = (gridDim.z > 1) ? (P + (long)blockIdx.z * sliceMN) : C;

    // ---- per-thread staging geometry (each thread: 2 float4 of A, 2 of B) ----
    const int aRow = t >> 3;             // 0..31  (rows aRow, aRow+32)
    const int aK   = (t & 7) << 2;       // 0,4,..,28
    const int bK   = t >> 4;             // non-trans: k rows bK, bK+16
    const int bN   = (t & 15) << 2;      // non-trans: 4 consecutive n
    const int tbN  = t >> 3;             // trans: n rows tbN, tbN+32
    const int tbK  = (t & 7) << 2;       // trans: 4 consecutive k

    const float* pA0 = A + (m0 + aRow) * (long)lda + kStart + aK;
    const float* pA1 = pA0 + 32 * (long)lda;
    const float* pB0;
    const float* pB1;
    long bStep;
    if (bTrans) { pB0 = B + (n0 + tbN) * (long)ldb + kStart + tbK; pB1 = pB0 + 32 * (long)ldb; bStep = TK; }
    else        { pB0 = B + (long)(kStart + bK) * ldb + n0 + bN;   pB1 = pB0 + 16 * (long)ldb; bStep = (long)TK * ldb; }

    // stage tile 0 into registers (global_load_b128)
    float4 a0 = *(const float4*)pA0;
    float4 a1 = *(const float4*)pA1;
    float4 b0 = *(const float4*)pB0;
    float4 b1 = *(const float4*)pB1;

    v8f acc[2] = {};

    for (int kk = 0; kk < kLen; kk += TK) {
        __syncthreads();   // all waves finished reading LDS of previous tile

        // ---- split current tile to hi/lo bf16 planes in LDS ----
        BF4 sa0 = split4(a0), sa1 = split4(a1);
        BF4 sb0 = split4(b0), sb1 = split4(b1);
        store4(&AsHi[aRow * TKP + aK], sa0.h);        store4(&AsLo[aRow * TKP + aK], sa0.l);
        store4(&AsHi[(aRow + 32) * TKP + aK], sa1.h); store4(&AsLo[(aRow + 32) * TKP + aK], sa1.l);
        if (bTrans) {
            store4(&BsHi[tbN * TKP + tbK], sb0.h);        store4(&BsLo[tbN * TKP + tbK], sb0.l);
            store4(&BsHi[(tbN + 32) * TKP + tbK], sb1.h); store4(&BsLo[(tbN + 32) * TKP + tbK], sb1.l);
        } else {
            #pragma unroll
            for (int jj = 0; jj < 4; ++jj) {           // transpose scatter [k][n] -> [n][k]
                BsHi[(bN + jj) * TKP + bK]      = sb0.h[jj];
                BsLo[(bN + jj) * TKP + bK]      = sb0.l[jj];
                BsHi[(bN + jj) * TKP + bK + 16] = sb1.h[jj];
                BsLo[(bN + jj) * TKP + bK + 16] = sb1.l[jj];
            }
        }
        __syncthreads();

        // ---- register-stage next tile (loads overlap with WMMA below) ----
        if (kk + TK < kLen) {
            pA0 += TK; pA1 += TK; pB0 += bStep; pB1 += bStep;
            a0 = *(const float4*)pA0;
            a1 = *(const float4*)pA1;
            b0 = *(const float4*)pB0;
            b1 = *(const float4*)pB1;
            if (kk + 2 * TK < kLen) {                  // gfx1250 global_prefetch, adj stream
                __builtin_prefetch(pA0 + TK, 0, 1);
                __builtin_prefetch(pA1 + TK, 0, 1);
            }
        }

        // ---- fragments: two aligned uint4 (ds_load_b128) each ----
        union Frag { v16bf v; uint4 q[2]; };
        Frag ahi, alo;
        {
            const int ab = (mt * 16 + lm) * TKP + 8 * lh;   // A 16x32 layout
            ahi.q[0] = *(const uint4*)&AsHi[ab];
            ahi.q[1] = *(const uint4*)&AsHi[ab + 16];
            alo.q[0] = *(const uint4*)&AsLo[ab];
            alo.q[1] = *(const uint4*)&AsLo[ab + 16];
        }
        #pragma unroll
        for (int ti = 0; ti < 2; ++ti) {
            const int nb = ((nt0 + ti) * 16 + lm) * TKP + 8 * lh;  // B 32x16 layout
            Frag bhi, blo;
            bhi.q[0] = *(const uint4*)&BsHi[nb];
            bhi.q[1] = *(const uint4*)&BsHi[nb + 16];
            blo.q[0] = *(const uint4*)&BsLo[nb];
            blo.q[1] = *(const uint4*)&BsLo[nb + 16];
            acc[ti] = __builtin_amdgcn_wmma_f32_16x16x32_bf16(false, ahi.v, false, bhi.v, (short)0, acc[ti], false, false);
            acc[ti] = __builtin_amdgcn_wmma_f32_16x16x32_bf16(false, ahi.v, false, blo.v, (short)0, acc[ti], false, false);
            acc[ti] = __builtin_amdgcn_wmma_f32_16x16x32_bf16(false, alo.v, false, bhi.v, (short)0, acc[ti], false, false);
        }
    }

    // ---- epilogue: C 16x16 layout -> VGPR r: M=r+8*lh, N=lm ----
    #pragma unroll
    for (int ti = 0; ti < 2; ++ti) {
        long col = n0 + (nt0 + ti) * 16 + lm;
        #pragma unroll
        for (int r = 0; r < 8; ++r) {
            long row = m0 + mt * 16 + r + 8 * lh;
            Ct[row * (long)ldc + col] = acc[ti][r];
        }
    }
}

// Fixed-order partial-sum reduction (deterministic), optional fused ReLU. float4-wide.
__global__ void reduce_partials(const float4* __restrict__ P, float4* __restrict__ out,
                                long MN4, int chunks, int doRelu)
{
    long i = (long)blockIdx.x * blockDim.x + threadIdx.x;
    if (i >= MN4) return;
    float4 s = make_float4(0.f, 0.f, 0.f, 0.f);
    for (int c = 0; c < chunks; ++c) {
        float4 p = P[(long)c * MN4 + i];
        s.x += p.x; s.y += p.y; s.z += p.z; s.w += p.w;
    }
    if (doRelu) {
        s.x = fmaxf(s.x, 0.f); s.y = fmaxf(s.y, 0.f);
        s.z = fmaxf(s.z, 0.f); s.w = fmaxf(s.w, 0.f);
    }
    out[i] = s;
}

// Bcat[k][0:64]=W2[k][:], Bcat[k][64:128]=W3[k][:]  (fuses mu/logvar adj-GEMMs)
__global__ void pack_bcat(const float4* __restrict__ W2, const float4* __restrict__ W3,
                          float4* __restrict__ Bc)
{
    int i = blockIdx.x * blockDim.x + threadIdx.x;   // over 64*128/4
    if (i >= 64 * 32) return;
    int k = i >> 5, c = i & 31;
    Bc[i] = (c < 16) ? W2[k * 16 + c] : W3[k * 16 + (c - 16)];
}

// T4 rows are [mu_i | logvar_i] (128 wide) -> separate mu / logvar outputs
__global__ void split_mu_lv(const float4* __restrict__ T4, float4* __restrict__ mu,
                            float4* __restrict__ lv, long NH4)
{
    long i = (long)blockIdx.x * blockDim.x + threadIdx.x;  // over N*64/4
    if (i >= NH4) return;
    long r = i >> 4; int j = (int)(i & 15);
    mu[i] = T4[r * 32 + j];
    lv[i] = T4[r * 32 + 16 + j];
}

extern "C" void kernel_launch(void* const* d_in, const int* in_sizes, int n_in,
                              void* d_out, int out_size, void* d_ws, size_t ws_size,
                              hipStream_t stream)
{
    (void)in_sizes; (void)n_in; (void)out_size; (void)ws_size;
    const float* x   = (const float*)d_in[0];   // [8192,256]
    const float* adj = (const float*)d_in[1];   // [8192,8192]
    const float* W1  = (const float*)d_in[2];   // [256,64]
    const float* W2  = (const float*)d_in[3];   // [64,64]
    const float* W3  = (const float*)d_in[4];   // [64,64]

    const long N = 8192, H = 64;
    float* out   = (float*)d_out;
    float* recon = out;                 // [N,N]
    float* mu_o  = out + N * N;         // [N,64]
    float* lv_o  = mu_o + N * H;        // [N,64]

    float* w   = (float*)d_ws;
    float* S1  = w;                      // N*64
    float* H1  = S1  + N * H;            // N*64
    float* Bc  = H1  + N * H;            // 64*128
    float* S23 = Bc  + H * 2 * H;        // N*128
    float* T4  = S23 + N * 2 * H;        // N*128
    float* P   = T4  + N * 2 * H;        // CH * N*128 partial slabs

    const int CH = 4;                    // deterministic K-split
    dim3 blk(256);

    // S1 = x @ W1                      (K=256, single chunk)
    gemm_bf16x3<<<dim3(N / 64, 1, 1), blk, 0, stream>>>(x, 256, W1, 64, 0, S1, 64, nullptr, 0, 256);
    // partials of adj @ S1             (K=8192, 4 chunks of 2048)
    gemm_bf16x3<<<dim3(N / 64, 1, CH), blk, 0, stream>>>(adj, 8192, S1, 64, 0, nullptr, 64, P, N * H, 8192 / CH);
    // H1 = relu(sum partials)
    reduce_partials<<<dim3((unsigned)((N * H / 4 + 255) / 256)), blk, 0, stream>>>(
        (const float4*)P, (float4*)H1, N * H / 4, CH, 1);
    // Bcat = [W2 | W3]
    pack_bcat<<<dim3((64 * 32 + 255) / 256), blk, 0, stream>>>(
        (const float4*)W2, (const float4*)W3, (float4*)Bc);
    // S23 = H1 @ Bcat                  (K=64, N=128)
    gemm_bf16x3<<<dim3(N / 64, 2, 1), blk, 0, stream>>>(H1, 64, Bc, 128, 0, S23, 128, nullptr, 0, 64);
    // partials of adj @ S23            (one adj pass computes mu AND logvar)
    gemm_bf16x3<<<dim3(N / 64, 2, CH), blk, 0, stream>>>(adj, 8192, S23, 128, 0, nullptr, 128, P, N * 2 * H, 8192 / CH);
    // T4 = sum partials  ( [mu | logvar] )
    reduce_partials<<<dim3((unsigned)((N * 2 * H / 4 + 255) / 256)), blk, 0, stream>>>(
        (const float4*)P, (float4*)T4, N * 2 * H / 4, CH, 0);
    // mu / logvar outputs
    split_mu_lv<<<dim3((unsigned)((N * H / 4 + 255) / 256)), blk, 0, stream>>>(
        (const float4*)T4, (float4*)mu_o, (float4*)lv_o, N * H / 4);
    // recon = mu @ mu^T  (A = mu cols of T4, lda=128; B transposed from T4; K=64)
    gemm_bf16x3<<<dim3(N / 64, N / 64, 1), blk, 0, stream>>>(T4, 128, T4, 128, 1, recon, 8192, nullptr, 0, 64);
}
